// cross_att_35407710388683
// MI455X (gfx1250) — matrix operand order repaired and checked by reading, not compiled, not run on hardware
//
#include <hip/hip_runtime.h>

typedef __bf16 bf16_t;
typedef __attribute__((ext_vector_type(16))) __bf16 v16bf;
typedef __attribute__((ext_vector_type(8)))  __bf16 v8bf;
typedef __attribute__((ext_vector_type(8)))  float   v8f;
typedef __attribute__((ext_vector_type(4)))  unsigned int v4u;
typedef __attribute__((ext_vector_type(8)))  int v8i;
typedef __attribute__((ext_vector_type(4)))  int v4i;
// GCC-style 16B int vector: matches the async-to-LDS builtin's pointee type.
typedef int v4i_gcc __attribute__((vector_size(16)));

static_assert(sizeof(bf16_t) == 2, "bf16 must be 2 bytes");

// -------- feature probes (compile-safe: fall back to sync staging) ----------
#if __has_builtin(__builtin_amdgcn_global_load_async_to_lds_b128)
#define HAVE_ASYNC_LDS 1
#else
#define HAVE_ASYNC_LDS 0
#endif

#if __has_builtin(__builtin_amdgcn_tensor_load_to_lds)
#if defined(__clang_major__) && (__clang_major__ >= 23)
#define HAVE_TDM6 1
#define HAVE_TDM5 0
#else
#define HAVE_TDM5 1
#define HAVE_TDM6 0
#endif
#else
#define HAVE_TDM5 0
#define HAVE_TDM6 0
#endif
#define HAVE_TDM (HAVE_TDM5 || HAVE_TDM6)

// Problem dims (fixed by the reference)
constexpr int B_    = 8;
constexpr int S_    = 2048;
constexpr int D_    = 1024;
constexpr int H_    = 8;
constexpr int MROWS = B_ * S_;   // 16384

// GEMM tiling: 128x128 block tile, K-step 32 (bf16 WMMA K), 256 threads = 8 waves.
// Wave grid 4(M) x 2(N): each wave owns a 32x64 sub-tile = 2x4 WMMA accumulators.
constexpr int BM  = 128;
constexpr int BN  = 128;
constexpr int BK  = 32;
constexpr int LDK = BK + 8;      // padded LDS row: 40 bf16 = 80B (16B aligned, conflict-free)
// TDM padding: 32 bf16 = 16 DWORDs data, then 8 bf16 = 4 DWORDs pad -> LDK stride.

__device__ __forceinline__ v16bf cat8(v8bf lo, v8bf hi) {
  return __builtin_shufflevector(lo, hi, 0,1,2,3,4,5,6,7,8,9,10,11,12,13,14,15);
}

__device__ __forceinline__ float wave_sum(float x) {
  #pragma unroll
  for (int o = 16; o > 0; o >>= 1) x += __shfl_xor(x, o, 32);
  return x;
}
__device__ __forceinline__ float wave_max(float x) {
  #pragma unroll
  for (int o = 16; o > 0; o >>= 1) x = fmaxf(x, __shfl_xor(x, o, 32));
  return x;
}

// 16-byte global -> LDS copy: async (ASYNCcnt-tracked) when available.
__device__ __forceinline__ void copy16_to_lds(bf16_t* ldst, const bf16_t* gsrc) {
#if HAVE_ASYNC_LDS
  __builtin_amdgcn_global_load_async_to_lds_b128(
      (__attribute__((address_space(1))) v4i_gcc*)gsrc,
      (__attribute__((address_space(3))) v4i_gcc*)ldst, 0, 0);
#else
  *(v8bf*)ldst = *(const v8bf*)gsrc;
#endif
}

template <int N>
__device__ __forceinline__ void wait_async_n() {
#if HAVE_ASYNC_LDS
#if __has_builtin(__builtin_amdgcn_s_wait_asynccnt)
  __builtin_amdgcn_s_wait_asynccnt(N);
#else
  asm volatile("s_wait_asynccnt %0" :: "i"(N) : "memory");
#endif
#endif
}

template <int N>
__device__ __forceinline__ void wait_tensor_n() {
#if __has_builtin(__builtin_amdgcn_s_wait_tensorcnt)
  __builtin_amdgcn_s_wait_tensorcnt(N);
#else
  asm volatile("s_wait_tensorcnt %0" :: "i"(N) : "memory");
#endif
}

#if HAVE_TDM
// TDM load of a 128-row x 32-bf16 tile (row stride 1024 bf16) into LDS with
// 16B padding after every 64B row -> matches bf16_t[128][LDK] layout.
// D# per CDNA5 ISA ch.8: group0 {count, lds_addr, global_addr, type=2},
// group1 {data_size=2B, pad_enable, pad_interval=16DW(code 3), pad_amount=4DW(code 3),
//         tensor_dim0=1024, tensor_dim1=big, tile_dim0=32, tile_dim1=128, stride0=1024}.
__device__ __forceinline__ void tdm_load_tile_128x32(unsigned lds_addr,
                                                     const bf16_t* gsrc) {
  const unsigned long long ga = (unsigned long long)gsrc;
  v4u g0;
  g0[0] = 1u;                                    // count=1, user descriptor
  g0[1] = lds_addr;                              // LDS byte address
  g0[2] = (unsigned)(ga & 0xFFFFFFFFull);        // global addr [31:0]
  g0[3] = (unsigned)((ga >> 32) & 0x1FFFFFFull)  // global addr [56:32]
        | 0x80000000u;                           // type=2 ("image") in [127:126]
  v8i g1;
  g1[0] = (1 << 16)        // data_size = 2 bytes
        | (1 << 20)        // pad_enable
        | (3 << 22)        // pad_interval: 16 DWORDs
        | (3 << 25);       // pad_amount: 4 DWORDs
  g1[1] = (int)((unsigned)D_ << 16);             // tensor_dim0 = 1024 (low 16 into [31:16])
  g1[2] = (int)(16384u << 16) | 0;               // dim0 hi=0, tensor_dim1 low16 = 16384
  g1[3] = (32 << 16) | 0;                        // tensor_dim1 hi=0, tile_dim0 = 32
  g1[4] = 128;                                   // tile_dim1 = 128 rows, tile_dim2 = 0
  g1[5] = D_;                                    // tensor_dim0_stride = 1024 (low 32)
  g1[6] = 0;
  g1[7] = 0;
  v4i gz;
  gz[0] = 0; gz[1] = 0; gz[2] = 0; gz[3] = 0;
#if HAVE_TDM5
  __builtin_amdgcn_tensor_load_to_lds(g0, g1, gz, gz, 0);
#else
  v8i gz8;
  #pragma unroll
  for (int i = 0; i < 8; ++i) gz8[i] = 0;
  __builtin_amdgcn_tensor_load_to_lds(g0, g1, gz, gz, gz8, 0);
#endif
}
#endif  // HAVE_TDM

// Load an A fragment (16x32 bf16) for wave-row `row16` from padded LDS tile.
// ISA layout: lanes 0-15 (M=lane): elems 0..7 = K0..7, elems 8..15 = K16..23;
//             lanes 16-31:         elems 0..7 = K8..15, elems 8..15 = K24..31.
__device__ __forceinline__ v16bf load_afrag(const bf16_t (*Asl)[LDK], int row16, int lane) {
  const int l16 = lane & 15, lh = lane >> 4;
  const int row = row16 + l16;
  const int koff = lh * 8;
  v8bf lo = *(const v8bf*)&Asl[row][koff];
  v8bf hi = *(const v8bf*)&Asl[row][koff + 16];
  return cat8(lo, hi);
}

// Load a B fragment (32x16 bf16) for wave-col `col16` from LDS tile stored as Bt[n][k].
// ISA layout: lane holds column N = lane&15; lanes 0-15: K0..7,K16..23; lanes 16-31: K8..15,K24..31.
__device__ __forceinline__ v16bf load_bfrag(const bf16_t (*Bsl)[LDK], int col16, int lane) {
  const int l16 = lane & 15, lh = lane >> 4;
  const int col = col16 + l16;
  const int kb  = lh * 8;
  v8bf lo = *(const v8bf*)&Bsl[col][kb];
  v8bf hi = *(const v8bf*)&Bsl[col][kb + 16];
  return cat8(lo, hi);
}

__device__ __forceinline__ void zero_acc(v8f acc[2][4]) {
  #pragma unroll
  for (int i = 0; i < 2; ++i)
    #pragma unroll
    for (int j = 0; j < 4; ++j)
      #pragma unroll
      for (int e = 0; e < 8; ++e) acc[i][j][e] = 0.0f;
}

// ---------------------------------------------------------------------------
// Kernel 1: projection GEMM.  Cout[row, col] = sum_k A[row,k]*W[k,col] + bias[col]
// A: fp32 [MROWS, D_], W: fp32 [D_, D_], Cout: bf16 [MROWS, D_]
// fp32 -> bf16 conversion happens while staging tiles into LDS (VALU path);
// the compiler software-pipelines the next-tile global loads across the WMMAs.
// ---------------------------------------------------------------------------
__global__ __launch_bounds__(256)
void proj_gemm_k(const float* __restrict__ A, const float* __restrict__ W,
                 const float* __restrict__ bias, bf16_t* __restrict__ Cout)
{
  __shared__ bf16_t Asl[BM][LDK];
  __shared__ bf16_t Bsl[BN][LDK];   // transposed: Bsl[n][k] = W[k0+k][n0+n]

  const int tid  = threadIdx.x;
  const int lane = tid & 31;
  const int wid  = tid >> 5;
  const int m0   = blockIdx.x * BM;
  const int n0   = blockIdx.y * BN;
  const int wm   = (wid >> 1) * 32;
  const int wn   = (wid & 1) * 64;

  v8f acc[2][4];
  zero_acc(acc);

  for (int k0 = 0; k0 < D_; k0 += BK) {
    // Stage A tile: 128x32 fp32 -> bf16. 1024 float4 chunks, 4 per thread.
    #pragma unroll
    for (int i = 0; i < 4; ++i) {
      const int fid = tid + i * 256;
      const int r   = fid >> 3;          // 8 float4 per row
      const int c4  = (fid & 7) * 4;
      const float4 f = *(const float4*)(A + (size_t)(m0 + r) * D_ + k0 + c4);
      bf16_t* dst = &Asl[r][c4];
      dst[0] = (bf16_t)f.x; dst[1] = (bf16_t)f.y;
      dst[2] = (bf16_t)f.z; dst[3] = (bf16_t)f.w;
    }
    // Stage W tile: 32x128 fp32 -> bf16, transposed into Bsl[n][k].
    #pragma unroll
    for (int i = 0; i < 4; ++i) {
      const int fid = tid + i * 256;
      const int kr  = fid >> 5;          // 32 float4 per row
      const int c4  = (fid & 31) * 4;
      const float4 f = *(const float4*)(W + (size_t)(k0 + kr) * D_ + n0 + c4);
      Bsl[c4 + 0][kr] = (bf16_t)f.x;
      Bsl[c4 + 1][kr] = (bf16_t)f.y;
      Bsl[c4 + 2][kr] = (bf16_t)f.z;
      Bsl[c4 + 3][kr] = (bf16_t)f.w;
    }
    if (k0 + BK < D_) {   // hint next tiles into cache (global_prefetch_b8)
      __builtin_prefetch(A + (size_t)(m0 + (tid >> 1)) * D_ + k0 + BK, 0, 1);
      __builtin_prefetch(W + (size_t)(k0 + BK + (tid & 31)) * D_ + n0, 0, 1);
    }
    __syncthreads();

    v16bf af[2], bf[4];
    #pragma unroll
    for (int mt = 0; mt < 2; ++mt) af[mt] = load_afrag(Asl, wm + mt * 16, lane);
    #pragma unroll
    for (int nt = 0; nt < 4; ++nt) bf[nt] = load_bfrag(Bsl, wn + nt * 16, lane);
    #pragma unroll
    for (int mt = 0; mt < 2; ++mt)
      #pragma unroll
      for (int nt = 0; nt < 4; ++nt)
        acc[mt][nt] = __builtin_amdgcn_wmma_f32_16x16x32_bf16(
            false, af[mt], false, bf[nt], (short)0, acc[mt][nt], false, false);
    __syncthreads();
  }

  // Epilogue: + bias, convert to bf16, store.
  const int l16 = lane & 15, lh = lane >> 4;
  #pragma unroll
  for (int mt = 0; mt < 2; ++mt)
    #pragma unroll
    for (int nt = 0; nt < 4; ++nt) {
      const int col = n0 + wn + nt * 16 + l16;
      const float bv = bias[col];
      #pragma unroll
      for (int e = 0; e < 8; ++e) {
        const int row = m0 + wm + mt * 16 + 8 * lh + e;
        Cout[(size_t)row * D_ + col] = (bf16_t)(acc[mt][nt][e] + bv);
      }
    }
}

// ---------------------------------------------------------------------------
// Kernel 2: scores = (Q @ K^T) / kd  per batch, bf16 in / bf16 out.
// Double-buffered: wave 0 issues TDM tensor_load_to_lds for tile i+1 into the
// ping buffer, then waits TENSORcnt<=2 (the two just-issued loads), so tile i
// is guaranteed resident while tile i+1 streams in behind the WMMAs.
// ---------------------------------------------------------------------------
__global__ __launch_bounds__(256)
void score_gemm_k(const bf16_t* __restrict__ Q, const bf16_t* __restrict__ Km,
                  bf16_t* __restrict__ P)
{
  __shared__ bf16_t Asl[2][BM][LDK];
  __shared__ bf16_t Bsl[2][BN][LDK];

  const int bz = blockIdx.z;
  Q  += (size_t)bz * S_ * D_;
  Km += (size_t)bz * S_ * D_;
  P  += (size_t)bz * S_ * S_;

  const int tid  = threadIdx.x;
  const int lane = tid & 31;
  const int wid  = tid >> 5;
  const int m0   = blockIdx.x * BM;   // query rows
  const int n0   = blockIdx.y * BN;   // key cols
  const int wm   = (wid >> 1) * 32;
  const int wn   = (wid & 1) * 64;

#if HAVE_TDM
  // LDS byte addresses (flat LDS aperture pointer truncates to LDS offset).
  const unsigned ldsA[2] = { (unsigned)(unsigned long long)&Asl[0][0][0],
                             (unsigned)(unsigned long long)&Asl[1][0][0] };
  const unsigned ldsB[2] = { (unsigned)(unsigned long long)&Bsl[0][0][0],
                             (unsigned)(unsigned long long)&Bsl[1][0][0] };
#endif

  auto stage = [&](int buf, int k0) {
#if HAVE_TDM
    if (wid == 0) {
      tdm_load_tile_128x32(ldsA[buf], Q  + (size_t)m0 * D_ + k0);
      tdm_load_tile_128x32(ldsB[buf], Km + (size_t)n0 * D_ + k0);
    }
#else
    #pragma unroll
    for (int i = 0; i < 2; ++i) {
      const int cid = tid + i * 256;
      const int r   = cid >> 2;
      const int c8  = (cid & 3) * 8;
      copy16_to_lds(&Asl[buf][r][c8], Q  + (size_t)(m0 + r) * D_ + k0 + c8);
      copy16_to_lds(&Bsl[buf][r][c8], Km + (size_t)(n0 + r) * D_ + k0 + c8);
    }
#endif
  };

  v8f acc[2][4];
  zero_acc(acc);

  constexpr int nIter = D_ / BK;
  stage(0, 0);
  for (int i = 0; i < nIter; ++i) {
    const int p  = i & 1;
    const int k0 = i * BK;
    if (i + 1 < nIter) {
      stage(1 - p, k0 + BK);            // prefetch next tile into ping buffer
#if HAVE_TDM
      if (wid == 0) wait_tensor_n<2>(); // older pair (tile i) has landed
#else
      wait_async_n<4>();                // 4 async loads per wave per stage
#endif
    } else {
#if HAVE_TDM
      if (wid == 0) wait_tensor_n<0>();
#else
      wait_async_n<0>();
#endif
    }
    __syncthreads();

    v16bf af[2], bf[4];
    #pragma unroll
    for (int mt = 0; mt < 2; ++mt) af[mt] = load_afrag(Asl[p], wm + mt * 16, lane);
    #pragma unroll
    for (int nt = 0; nt < 4; ++nt) bf[nt] = load_bfrag(Bsl[p], wn + nt * 16, lane);
    #pragma unroll
    for (int mt = 0; mt < 2; ++mt)
      #pragma unroll
      for (int nt = 0; nt < 4; ++nt)
        acc[mt][nt] = __builtin_amdgcn_wmma_f32_16x16x32_bf16(
            false, af[mt], false, bf[nt], (short)0, acc[mt][nt], false, false);
    __syncthreads();   // all reads of buffer p done before it is re-staged
  }

  const float kdinv = rsqrtf((float)S_ / (float)H_);   // 1/16
  const int l16 = lane & 15, lh = lane >> 4;
  #pragma unroll
  for (int mt = 0; mt < 2; ++mt)
    #pragma unroll
    for (int nt = 0; nt < 4; ++nt) {
      const int col = n0 + wn + nt * 16 + l16;
      #pragma unroll
      for (int e = 0; e < 8; ++e) {
        const int row = m0 + wm + mt * 16 + 8 * lh + e;
        P[(size_t)row * S_ + col] = (bf16_t)(acc[mt][nt][e] * kdinv);
      }
    }
}

// ---------------------------------------------------------------------------
// Kernel 3: in-place row softmax over S_=2048 bf16 values. One block per row.
// ---------------------------------------------------------------------------
__global__ __launch_bounds__(256)
void softmax_rows_k(bf16_t* __restrict__ P)
{
  const size_t row = blockIdx.x;
  bf16_t* p = P + row * (size_t)S_;
  const int tid = threadIdx.x, lane = tid & 31, wid = tid >> 5;
  __shared__ float red[8];

  v8bf raw = *(const v8bf*)(p + tid * 8);
  float v[8];
  float m = -3.0e38f;
  #pragma unroll
  for (int e = 0; e < 8; ++e) { v[e] = (float)raw[e]; m = fmaxf(m, v[e]); }
  m = wave_max(m);
  if (lane == 0) red[wid] = m;
  __syncthreads();
  float bm = red[0];
  #pragma unroll
  for (int w = 1; w < 8; ++w) bm = fmaxf(bm, red[w]);

  float s = 0.f;
  #pragma unroll
  for (int e = 0; e < 8; ++e) { v[e] = __expf(v[e] - bm); s += v[e]; }
  s = wave_sum(s);
  __syncthreads();
  if (lane == 0) red[wid] = s;
  __syncthreads();
  float bs = 0.f;
  #pragma unroll
  for (int w = 0; w < 8; ++w) bs += red[w];
  const float inv = 1.0f / bs;

  v8bf o;
  #pragma unroll
  for (int e = 0; e < 8; ++e) o[e] = (bf16_t)(v[e] * inv);
  *(v8bf*)(p + tid * 8) = o;
}

// ---------------------------------------------------------------------------
// Kernel 4: line = P @ V, fused epilogue: RES = line*sum(w1) + b1 + x (fp32).
// Double-buffered: A-tile via async global->LDS b128 (s_wait_asynccnt 2 keeps
// the next tile in flight behind the WMMAs); B-tile transposed on the ds path.
// ---------------------------------------------------------------------------
__global__ __launch_bounds__(256)
void line_gemm_k(const bf16_t* __restrict__ P, const bf16_t* __restrict__ V,
                 const float* __restrict__ X, const float* __restrict__ w1,
                 const float* __restrict__ b1, float* __restrict__ RES)
{
  __shared__ bf16_t Asl[2][BM][LDK];
  __shared__ bf16_t Bsl[2][BN][LDK];   // Bsl[buf][n][k] = V[k0+k][n0+n]

  const int bz = blockIdx.z;
  P += (size_t)bz * S_ * S_;
  V += (size_t)bz * S_ * D_;

  const int tid  = threadIdx.x;
  const int lane = tid & 31;
  const int wid  = tid >> 5;
  const int m0   = blockIdx.x * BM;   // query rows (in batch)
  const int n0   = blockIdx.y * BN;   // feature cols
  const int wm   = (wid >> 1) * 32;
  const int wn   = (wid & 1) * 64;

  auto stage = [&](int buf, int k0) {
    // A tile: 128 rows x 32 keys of P (contiguous bf16 rows) - async copies.
    #pragma unroll
    for (int i = 0; i < 2; ++i) {
      const int cid = tid + i * 256;
      const int r   = cid >> 2;
      const int c8  = (cid & 3) * 8;
      copy16_to_lds(&Asl[buf][r][c8], P + (size_t)(m0 + r) * S_ + k0 + c8);
    }
    // B tile: 32 key-rows x 128 cols of V, transposed into Bsl[buf][n][k].
    #pragma unroll
    for (int i = 0; i < 2; ++i) {
      const int cid = tid + i * 256;
      const int kr  = cid >> 4;          // 16 chunks per key row
      const int c8  = (cid & 15) * 8;
      v8bf chunk = *(const v8bf*)(V + (size_t)(k0 + kr) * D_ + n0 + c8);
      #pragma unroll
      for (int e = 0; e < 8; ++e) Bsl[buf][c8 + e][kr] = chunk[e];
    }
  };

  v8f acc[2][4];
  zero_acc(acc);

  constexpr int nIter = S_ / BK;
  stage(0, 0);
  for (int i = 0; i < nIter; ++i) {
    const int p  = i & 1;
    const int k0 = i * BK;
    if (i + 1 < nIter) {
      stage(1 - p, k0 + BK);   // prefetch next tile into ping buffer
      wait_async_n<2>();       // older async pair (tile i's A) has landed
    } else {
      wait_async_n<0>();
    }
    __syncthreads();

    v16bf af[2], bf[4];
    #pragma unroll
    for (int mt = 0; mt < 2; ++mt) af[mt] = load_afrag(Asl[p], wm + mt * 16, lane);
    #pragma unroll
    for (int nt = 0; nt < 4; ++nt) bf[nt] = load_bfrag(Bsl[p], wn + nt * 16, lane);
    #pragma unroll
    for (int mt = 0; mt < 2; ++mt)
      #pragma unroll
      for (int nt = 0; nt < 4; ++nt)
        acc[mt][nt] = __builtin_amdgcn_wmma_f32_16x16x32_bf16(
            false, af[mt], false, bf[nt], (short)0, acc[mt][nt], false, false);
    __syncthreads();   // all reads of buffer p done before it is re-staged
  }

  // Fused epilogue: line*sum(w1) + b1 + x  (residual), fp32 out.
  float sw = 0.f;
  #pragma unroll
  for (int h = 0; h < H_; ++h) sw += w1[h];
  const float bb = b1[0];

  const int l16 = lane & 15, lh = lane >> 4;
  #pragma unroll
  for (int mt = 0; mt < 2; ++mt)
    #pragma unroll
    for (int nt = 0; nt < 4; ++nt) {
      const int col = n0 + wn + nt * 16 + l16;
      #pragma unroll
      for (int e = 0; e < 8; ++e) {
        const int row = m0 + wm + mt * 16 + 8 * lh + e;
        const size_t g = ((size_t)bz * S_ + row) * (size_t)D_ + col;
        RES[g] = acc[mt][nt][e] * sw + bb + X[g];
      }
    }
}

// ---------------------------------------------------------------------------
// Kernel 5: LayerNorm over D_=1024 per row, fp32 in -> fp32 out.
// ---------------------------------------------------------------------------
__global__ __launch_bounds__(256)
void layernorm_k(const float* __restrict__ R, const float* __restrict__ gamma,
                 const float* __restrict__ beta, float* __restrict__ out)
{
  const size_t row = blockIdx.x;
  const float* r = R + row * (size_t)D_;
  const int tid = threadIdx.x, lane = tid & 31, wid = tid >> 5;
  __shared__ float s1[8], s2[8];

  const float4 f = *(const float4*)(r + tid * 4);
  float sum = f.x + f.y + f.z + f.w;
  float sq  = f.x * f.x + f.y * f.y + f.z * f.z + f.w * f.w;
  sum = wave_sum(sum);
  sq  = wave_sum(sq);
  if (lane == 0) { s1[wid] = sum; s2[wid] = sq; }
  __syncthreads();
  float ts = 0.f, tq = 0.f;
  #pragma unroll
  for (int w = 0; w < 8; ++w) { ts += s1[w]; tq += s2[w]; }
  const float mu  = ts * (1.0f / D_);
  const float var = tq * (1.0f / D_) - mu * mu;
  const float rs  = rsqrtf(var + 1e-5f);

  const float4 g  = *(const float4*)(gamma + tid * 4);
  const float4 be = *(const float4*)(beta + tid * 4);
  float4 o;
  o.x = (f.x - mu) * rs * g.x + be.x;
  o.y = (f.y - mu) * rs * g.y + be.y;
  o.z = (f.z - mu) * rs * g.z + be.z;
  o.w = (f.w - mu) * rs * g.w + be.w;
  *(float4*)(out + row * (size_t)D_ + tid * 4) = o;
}

// ---------------------------------------------------------------------------
extern "C" void kernel_launch(void* const* d_in, const int* in_sizes, int n_in,
                              void* d_out, int out_size, void* d_ws, size_t ws_size,
                              hipStream_t stream)
{
  const float* x     = (const float*)d_in[0];
  const float* x2    = (const float*)d_in[1];
  const float* Wq    = (const float*)d_in[2];
  const float* bq    = (const float*)d_in[3];
  const float* Wk    = (const float*)d_in[4];
  const float* bk    = (const float*)d_in[5];
  const float* Wv    = (const float*)d_in[6];
  const float* bv    = (const float*)d_in[7];
  const float* w1    = (const float*)d_in[8];
  const float* b1    = (const float*)d_in[9];
  const float* gamma = (const float*)d_in[10];
  const float* beta  = (const float*)d_in[11];
  float* out = (float*)d_out;

  // Workspace carve-up (~235 MB total).
  char* ws = (char*)d_ws;
  size_t off = 0;
  bf16_t* Qb = (bf16_t*)(ws + off); off += (size_t)MROWS * D_ * sizeof(bf16_t);
  bf16_t* Kb = (bf16_t*)(ws + off); off += (size_t)MROWS * D_ * sizeof(bf16_t);
  bf16_t* Vb = (bf16_t*)(ws + off); off += (size_t)MROWS * D_ * sizeof(bf16_t);
  bf16_t* Pb = (bf16_t*)(ws + off); off += (size_t)B_ * S_ * S_ * sizeof(bf16_t);
  float*  Rb = (float*) (ws + off); off += (size_t)MROWS * D_ * sizeof(float);

  const dim3 blk(256);
  proj_gemm_k<<<dim3(MROWS / BM, D_ / BN, 1), blk, 0, stream>>>(x2, Wq, bq, Qb);
  proj_gemm_k<<<dim3(MROWS / BM, D_ / BN, 1), blk, 0, stream>>>(x,  Wk, bk, Kb);
  proj_gemm_k<<<dim3(MROWS / BM, D_ / BN, 1), blk, 0, stream>>>(x,  Wv, bv, Vb);
  score_gemm_k<<<dim3(S_ / BM, S_ / BN, B_), blk, 0, stream>>>(Qb, Kb, Pb);
  softmax_rows_k<<<dim3(MROWS), blk, 0, stream>>>(Pb);
  line_gemm_k<<<dim3(S_ / BM, D_ / BN, B_), blk, 0, stream>>>(Pb, Vb, x, w1, b1, Rb);
  layernorm_k<<<dim3(MROWS), blk, 0, stream>>>(Rb, gamma, beta, out);
}
